// GATLayer_46961172415198
// MI455X (gfx1250) — matrix operand order repaired and checked by reading, not compiled, not run on hardware
//
#include <hip/hip_runtime.h>
#include <hip/hip_bf16.h>
#include <math.h>

typedef __attribute__((ext_vector_type(16))) _Float16 v16h;
typedef __attribute__((ext_vector_type(8)))  float    v8f;
typedef __attribute__((ext_vector_type(4)))  int      v4i;
typedef __attribute__((address_space(1))) v4i* v4i_g;   // global int4*
typedef __attribute__((address_space(3))) v4i* v4i_l;   // LDS int4*

#define B_SZ 4
#define N_SZ 2048
#define DIN  128
#define H_SZ 4
#define HD_SZ 32
#define NEG_SLOPE 0.2f
#define NW 4                       // waves per block in attention kernel
#define NWORDS (N_SZ / 32)         // packed mask words per row

#if __has_builtin(__builtin_amdgcn_global_load_async_to_lds_b128) && \
    __has_builtin(__builtin_amdgcn_s_wait_asynccnt)
#define USE_ASYNC_LDS 1
#else
#define USE_ASYNC_LDS 0
#endif

// ---------- wave32 WMMA f16 16x16x32 fragment index helpers (ISA 7.12.2) ----------
// A (16x32, MxK): lanes 0-15 -> M=lane, k-base 0; lanes 16-31 -> M=lane-16, k-base 16.
// half t (0..15): K = kbase + t (matches B's contiguous k order)
__device__ __forceinline__ int a_k_of(int lane, int t) {
  int kb = (lane >> 4) << 4;
  return kb + t;
}
// B (32x16, KxN): lanes 0-15 -> N=lane, K=0..15 (K=t); lanes 16-31 -> N=lane-16, K=16..31
// C/D (16x16 f32): vgpr r -> row = (lane>=16 ? 8:0)+r, col = lane&15

// ============ Kernel 0: bit-pack adjacency (16MB int32 -> 512KB bits) ============
__global__ void __launch_bounds__(256)
gat_pack_mask_kernel(const int* __restrict__ adj, unsigned* __restrict__ amask) {
  const int lane = threadIdx.x & 31;
  const int gw = blockIdx.x * 8 + (threadIdx.x >> 5);
  const int nw = gridDim.x * 8;
  const int words = (N_SZ * N_SZ) / 32;
  for (int w = gw; w < words; w += nw) {
    bool pred = adj[(size_t)w * 32 + lane] != 0;   // coalesced 128B per wave
#if __has_builtin(__builtin_amdgcn_ballot_w32)
    unsigned bal = __builtin_amdgcn_ballot_w32(pred);
#else
    unsigned bal = (unsigned)__ballot(pred);
#endif
    if (lane == 0) amask[w] = bal;
  }
}

// ================= Kernel 1: h = x @ W^T  (f32 in, f16 out), WMMA =================
__global__ void __launch_bounds__(128)
gat_proj_kernel(const float* __restrict__ x, const float* __restrict__ W,
                _Float16* __restrict__ h16) {
  const int lane = threadIdx.x & 31;
  const int wave = threadIdx.x >> 5;
  const int row0 = blockIdx.x * 16;                 // global row in [0, B*N)
  const int n0   = (blockIdx.y * 4 + wave) * 16;    // output-col tile
  const int m    = lane & 15;

  v8f c = {};
  const float* xrow = x + (size_t)(row0 + m) * DIN;
  const int kb_b = (lane >> 4) << 4;                // 0 or 16 (B k-base)

#pragma unroll
  for (int k0 = 0; k0 < DIN; k0 += 32) {
    v16h a, b;
#pragma unroll
    for (int t = 0; t < 16; ++t)
      a[t] = (_Float16)xrow[k0 + a_k_of(lane, t)];
    // B(k,n) = W[n, k]  (so that out = x @ W^T); contiguous in t
    const float* wrow = W + (size_t)(n0 + m) * DIN + k0 + kb_b;
#pragma unroll
    for (int t = 0; t < 16; ++t)
      b[t] = (_Float16)wrow[t];
    c = __builtin_amdgcn_wmma_f32_16x16x32_f16(false, a, false, b, (short)0, c,
                                               false, false);
  }
  const int rb = (lane >> 4) << 3;
#pragma unroll
  for (int r = 0; r < 8; ++r)
    h16[(size_t)(row0 + rb + r) * DIN + n0 + m] = (_Float16)c[r];
}

// ======= Kernel 2: ei/ej = <h, a1/a2> per head, stored transposed per (b,head) =======
__global__ void __launch_bounds__(256)
gat_ei_ej_kernel(const _Float16* __restrict__ h16, const float* __restrict__ a,
                 float* __restrict__ eiT, float* __restrict__ ejT) {
  int idx = blockIdx.x * blockDim.x + threadIdx.x;  // b*N + n
  if (idx >= B_SZ * N_SZ) return;
  const int b = idx / N_SZ, n = idx % N_SZ;
  const _Float16* hrow = h16 + (size_t)idx * DIN;
#pragma unroll
  for (int hh = 0; hh < H_SZ; ++hh) {
    float s1 = 0.f, s2 = 0.f;
#pragma unroll
    for (int d = 0; d < HD_SZ; ++d) {
      float v = (float)hrow[hh * HD_SZ + d];
      s1 += v * a[hh * 2 * HD_SZ + d];
      s2 += v * a[hh * 2 * HD_SZ + HD_SZ + d];
    }
    eiT[((size_t)b * H_SZ + hh) * N_SZ + n] = s1;
    ejT[((size_t)b * H_SZ + hh) * N_SZ + n] = s2;
  }
}

// ====== Kernel 3: fused masked-softmax attention + aggregation (flash-style) ======
// 4 waves/block; each wave owns one 16-row i-tile; the block shares LDS-staged
// V chunks (32 j-rows x 32 dims f16) and ej chunks for its (b, head).
__global__ void __launch_bounds__(128)
gat_attn_kernel(const _Float16* __restrict__ h16, const unsigned* __restrict__ amask,
                const float* __restrict__ eiT, const float* __restrict__ ejT,
                float* __restrict__ out) {
  __shared__ _Float16 s_v[32 * 32];     // V chunk (2KB)
  __shared__ float s_ej[32];
  __shared__ float s_scale[NW][16];
  __shared__ float s_l[NW][16];

  const int tid  = threadIdx.x;
  const int lane = tid & 31;
  const int wave = tid >> 5;
  const int head = blockIdx.y;
  const int b    = blockIdx.z;
  const int i0   = (blockIdx.x * NW + wave) * 16;
  const int bh   = b * H_SZ + head;

  const int m    = lane & 15;           // A-frag row / B,C-frag col
  const int kb_b = (lane >> 4) << 4;    // B-frag k-base (0/16)
  const int rb   = (lane >> 4) << 3;    // C-frag row-base (0/8)

  const float ei_m = eiT[(size_t)bh * N_SZ + i0 + m];
  const unsigned* mrow = amask + (size_t)(i0 + m) * NWORDS;
  const _Float16* hb = h16 + (size_t)b * N_SZ * DIN + head * HD_SZ;
  const float* ejb = ejT + (size_t)bh * N_SZ;

  // this thread's staging slice: one aligned 16B segment of the 32x32 V chunk
  const int vrow = tid >> 2, vseg = (tid & 3) * 8;

  float mrun = -INFINITY, lrun = 0.f;
  v8f acc0 = {}, acc1 = {};

  for (int j0 = 0; j0 < N_SZ; j0 += 32) {
    // ---- stage V chunk + ej chunk into LDS ----
    const _Float16* vsrc = hb + (size_t)(j0 + vrow) * DIN + vseg;
    __builtin_prefetch(hb + (size_t)(j0 + 32 + vrow) * DIN + vseg, 0, 1);
#if USE_ASYNC_LDS
    // signature: (v4i addrspace(1)* src, v4i addrspace(3)* dst, imm offset, imm cpol)
    __builtin_amdgcn_global_load_async_to_lds_b128(
        (v4i_g)(_Float16*)vsrc,
        (v4i_l)&s_v[vrow * 32 + vseg], 0, 0);
    if (tid < 32) s_ej[tid] = ejb[j0 + tid];
    __builtin_amdgcn_s_wait_asynccnt(0);
    __syncthreads();
#else
    *(uint4*)&s_v[vrow * 32 + vseg] = *(const uint4*)vsrc;
    if (tid < 32) s_ej[tid] = ejb[j0 + tid];
    __syncthreads();
#endif

    // ---- scores for this lane's 16 (m, k) A-fragment slots ----
    const unsigned bits = mrow[j0 >> 5];   // 32 adjacency bits for this row/chunk
    float p[16];
    float cmax = -INFINITY;
#pragma unroll
    for (int t = 0; t < 16; ++t) {
      int k = a_k_of(lane, t);
      float s = ei_m + s_ej[k];
      s = (s > 0.f) ? s : NEG_SLOPE * s;                  // leaky relu
      s = ((bits >> k) & 1u) ? s : -INFINITY;             // adjacency mask
      p[t] = s;
      cmax = fmaxf(cmax, s);
    }
    cmax = fmaxf(cmax, __shfl_xor(cmax, 16, 32));         // full 32-j row max
    float mnew  = fmaxf(mrun, cmax);
    float scale = (mnew == -INFINITY) ? 1.f : __expf(mrun - mnew);

    // ---- P = exp(s - mnew), build A fragment, row sums ----
    v16h af;
    float psum = 0.f;
#pragma unroll
    for (int t = 0; t < 16; ++t) {
      float pp = (p[t] == -INFINITY) ? 0.f : __expf(p[t] - mnew);
      psum += pp;
      af[t] = (_Float16)pp;
    }
    psum += __shfl_xor(psum, 16, 32);
    lrun = lrun * scale + psum;
    mrun = mnew;

    // broadcast per-row rescale factor to C-fragment lanes (wave-local LDS)
    if (lane < 16) s_scale[wave][m] = scale;
#pragma unroll
    for (int r = 0; r < 8; ++r) {
      float sc = s_scale[wave][rb + r];
      acc0[r] *= sc;
      acc1[r] *= sc;
    }

    // ---- B-fragments gathered from LDS: V(k,n) = s_v[k*32 + n] ----
    const _Float16* vcol = &s_v[kb_b * 32 + m];
    v16h b0, b1;
#pragma unroll
    for (int t = 0; t < 16; ++t) {
      b0[t] = vcol[t * 32];
      b1[t] = vcol[t * 32 + 16];
    }
    acc0 = __builtin_amdgcn_wmma_f32_16x16x32_f16(false, af, false, b0, (short)0,
                                                  acc0, false, false);
    acc1 = __builtin_amdgcn_wmma_f32_16x16x32_f16(false, af, false, b1, (short)0,
                                                  acc1, false, false);
    __syncthreads();   // protect s_v/s_ej before next chunk's staging
  }

  // ---- normalize by row sums and store out[b, i, head*32 + d] (fp32) ----
  if (lane < 16) s_l[wave][m] = lrun;
  float* orow = out + ((size_t)b * N_SZ + i0) * (H_SZ * HD_SZ) + head * HD_SZ;
#pragma unroll
  for (int r = 0; r < 8; ++r) {
    float inv = 1.0f / s_l[wave][rb + r];
    orow[(size_t)(rb + r) * (H_SZ * HD_SZ) + m]      = acc0[r] * inv;
    orow[(size_t)(rb + r) * (H_SZ * HD_SZ) + 16 + m] = acc1[r] * inv;
  }
}

// =============================== launcher ===============================
extern "C" void kernel_launch(void* const* d_in, const int* in_sizes, int n_in,
                              void* d_out, int out_size, void* d_ws, size_t ws_size,
                              hipStream_t stream) {
  const float* x   = (const float*)d_in[0];
  const int*   adj = (const int*)d_in[1];
  const float* W   = (const float*)d_in[2];
  const float* a   = (const float*)d_in[3];
  float* out = (float*)d_out;

  // workspace: h16 (2MB f16) | eiT (128KB) | ejT (128KB) | amask (512KB)
  _Float16* h16 = (_Float16*)d_ws;
  float* eiT = (float*)((char*)d_ws + (size_t)B_SZ * N_SZ * DIN * sizeof(_Float16));
  float* ejT = eiT + (size_t)B_SZ * N_SZ * H_SZ;
  unsigned* amask = (unsigned*)(ejT + (size_t)B_SZ * N_SZ * H_SZ);

  // 0) bit-pack adjacency
  gat_pack_mask_kernel<<<dim3(512), dim3(256), 0, stream>>>(adj, amask);
  // 1) projection GEMM: 512 row-tiles x 8 col-tiles (4 waves/block)
  gat_proj_kernel<<<dim3(B_SZ * N_SZ / 16, DIN / 64), dim3(128), 0, stream>>>(x, W, h16);
  // 2) attention logit halves (transposed layouts)
  gat_ei_ej_kernel<<<dim3((B_SZ * N_SZ + 255) / 256), dim3(256), 0, stream>>>(h16, a, eiT, ejT);
  // 3) fused masked softmax + aggregation
  gat_attn_kernel<<<dim3(N_SZ / (16 * NW), H_SZ, B_SZ), dim3(128), 0, stream>>>(
      h16, amask, eiT, ejT, out);
}
